// AverageCrossStitch_11931419148821
// MI455X (gfx1250) — matrix operand order repaired and checked by reading, not compiled, not run on hardware
//
#include <hip/hip_runtime.h>
#include <stdint.h>

typedef __attribute__((ext_vector_type(16))) __bf16 v16bf;
typedef __attribute__((ext_vector_type(8)))  float  v8f;
typedef __attribute__((ext_vector_type(4)))  unsigned int v4u;
typedef __attribute__((ext_vector_type(8)))  int    v8i;
typedef __attribute__((ext_vector_type(4)))  int    v4i;

#define BQ        32      // query rows per workgroup
#define BK        32      // kv rows per iteration
#define DIM       1024
#define SEQ       2048
#define BATCH     8
#define NTHREADS  512     // 16 wave32
#define DSLICE    64      // PV-phase D-slice per wave (DIM / 16 waves)
#define SDSLICE   256     // S-phase D-slice per wave group (DIM / 4 groups)

#if defined(__has_builtin)
#  if __has_builtin(__builtin_amdgcn_tensor_load_to_lds)
#    define HAVE_TDM 1
#  endif
#endif
#ifndef HAVE_TDM
#  define HAVE_TDM 0
#endif
#if __has_include(<hip/amd_detail/amd_gfx1250_TDM.h>)
#  define TDM_6ARG 1
#else
#  define TDM_6ARG 0
#endif

// ---------- bf16 helpers (round-to-nearest-even) ----------
__device__ __forceinline__ uint32_t pack_bf16(float a, float b) {
  uint32_t ua = __float_as_uint(a), ub = __float_as_uint(b);
  ua = (ua + 0x7FFFu + ((ua >> 16) & 1u)) >> 16;
  ub = (ub + 0x7FFFu + ((ub >> 16) & 1u)) >> 16;
  return ua | (ub << 16);
}

// ---------- WMMA fragment loaders (contiguous k = 16*(lane/16) + s for A and B) ----------
__device__ __forceinline__ v16bf load_a_frag(const unsigned short* p, int stride, int lane) {
  union { v16bf v; uint32_t u[8]; } f;
  const int m  = lane & 15;
  const int kh = (lane >> 4) << 4;
  const unsigned short* row = p + m * stride;
  #pragma unroll
  for (int v = 0; v < 8; ++v) {
    const int k = (v << 1) + kh;
    f.u[v] = *(const uint32_t*)(row + k);
  }
  return f.v;
}
__device__ __forceinline__ v16bf load_b_trans(const unsigned short* p, int stride, int lane) {
  union { v16bf v; uint32_t u[8]; } f;
  const int n  = lane & 15;
  const int k0 = (lane >> 4) << 4;
  const unsigned short* row = p + n * stride + k0;
  #pragma unroll
  for (int v = 0; v < 8; ++v)
    f.u[v] = *(const uint32_t*)(row + 2 * v);
  return f.v;
}
__device__ __forceinline__ v16bf load_b_direct(const unsigned short* p, int stride, int lane) {
  union { v16bf v; uint32_t u[8]; } f;
  const int n  = lane & 15;
  const int k0 = (lane >> 4) << 4;
  #pragma unroll
  for (int v = 0; v < 8; ++v) {
    const int k = k0 + 2 * v;
    uint32_t lo = p[(k)     * stride + n];
    uint32_t hi = p[(k + 1) * stride + n];
    f.u[v] = lo | (hi << 16);
  }
  return f.v;
}
__device__ __forceinline__ v8f wmma_bf16(v16bf a, v16bf b, v8f c) {
  return __builtin_amdgcn_wmma_f32_16x16x32_bf16(false, a, false, b, (short)0, c, false, false);
}

#if HAVE_TDM
__device__ __forceinline__ uint32_t lds_off(const void* p) {
  return (uint32_t)(uintptr_t)(__attribute__((address_space(3))) const void*)p;
}
// One TDM op: load a rows x rowlen tile of 2-byte elements (row-major, stride=rowlen)
// from global to LDS.  D# layout per CDNA5 ISA 08_async_tensor section 8.
__device__ __forceinline__ void tdm_load_tile(const void* gsrc, const void* ldst,
                                              int rows, int rowlen) {
  const uint64_t ga = (uint64_t)(uintptr_t)gsrc;
  v4u g0;
  g0[0] = 1u;                                               // count=1, user descriptor
  g0[1] = lds_off(ldst);                                    // lds_addr (bytes)
  g0[2] = (uint32_t)(ga & 0xFFFFFFFFu);                     // global_addr[31:0]
  g0[3] = (uint32_t)((ga >> 32) & 0x01FFFFFFu) | (2u << 30);// global_addr[56:32] | type=2
  v8i g1;
  g1[0] = (int)(1u << 16);                                  // wg_mask=0, data_size=1 (2B)
  g1[1] = (int)(((uint32_t)rowlen & 0xFFFFu) << 16);        // tensor_dim0[15:0]
  g1[2] = (int)(((uint32_t)rowlen >> 16) |
                (((uint32_t)rows & 0xFFFFu) << 16));        // tensor_dim0[31:16] | tensor_dim1[15:0]
  g1[3] = (int)(((uint32_t)rows >> 16) |
                ((uint32_t)rowlen << 16));                  // tensor_dim1[31:16] | tile_dim0
  g1[4] = rows;                                             // tile_dim1 (tile_dim2=0)
  g1[5] = rowlen;                                           // tensor_dim0_stride[31:0]
  g1[6] = 0;
  g1[7] = 0;
  v4i z4 = {0, 0, 0, 0};
#if TDM_6ARG
  v8i z8 = {0, 0, 0, 0, 0, 0, 0, 0};
  __builtin_amdgcn_tensor_load_to_lds(g0, g1, z4, z4, z8, 0);
#else
  __builtin_amdgcn_tensor_load_to_lds(g0, g1, z4, z4, 0);
#endif
}
#endif  // HAVE_TDM

// ---------------------------------------------------------------------------
// fp32 -> bf16 pre-conversion (memory bound, one shot)
// ---------------------------------------------------------------------------
__global__ __launch_bounds__(256)
void convert_bf16(const float* __restrict__ in, unsigned short* __restrict__ out, int n4) {
  const float4* src = (const float4*)in;
  uint32_t* dst = (uint32_t*)out;
  for (int i = blockIdx.x * blockDim.x + threadIdx.x; i < n4; i += gridDim.x * blockDim.x) {
    float4 x = src[i];
    dst[2 * i]     = pack_bf16(x.x, x.y);
    dst[2 * i + 1] = pack_bf16(x.z, x.w);
  }
}

// ---------------------------------------------------------------------------
// Flash-style cross attention: Out = 0.5*Q + 0.5*softmax(Q K^T / 32) * V, V==K.
// BF16IN: stream pre-converted bf16 tiles via TDM; else convert fp32 inline.
// ---------------------------------------------------------------------------
template <bool BF16IN>
__global__ __launch_bounds__(NTHREADS, 1)
void cross_attn(const unsigned short* __restrict__ Qb, const unsigned short* __restrict__ Kb,
                const float* __restrict__ Qf, const float* __restrict__ Kf,
                float* __restrict__ Og)
{
  __shared__ unsigned short sQ [BQ * DIM];   // 64 KB bf16 query tile (resident)
  __shared__ unsigned short sKV[BK * DIM];   // 64 KB bf16 key/value tile (streamed)
  __shared__ float          sS [BQ * BK];
  __shared__ unsigned short sP [BQ * BK];
  __shared__ float          sM [BQ];
  __shared__ float          sL [BQ];
  __shared__ float          sC [BQ];

  const int tid  = threadIdx.x;
  const int lane = tid & 31;
  const int wave = tid >> 5;
  const int bb   = blockIdx.y;
  const int m0   = blockIdx.x * BQ;
  const int ds0  = wave * DSLICE;            // PV-phase D-slice
  const int cIn  = lane & 15;
  const int hh   = (lane >> 4) << 3;
  // S-phase work split: 4 groups of 4 waves; each wave owns one 16x16 quadrant
  // of the 32x32 score tile over a 256-wide D-slice (4x fewer ds atomics).
  const int sQuadM = ((wave & 3) >> 1) * 16;
  const int sQuadN = (wave & 1) * 16;
  const int sDs0   = (wave >> 2) * SDSLICE;

  // ---- load Q tile once ----
  if (BF16IN) {
#if HAVE_TDM
    if (wave == 0) {
      tdm_load_tile(Qb + ((size_t)bb * SEQ + m0) * DIM, sQ, BQ, DIM);
      __builtin_amdgcn_s_wait_tensorcnt(0);
    }
#else
    const uint4* src = (const uint4*)(Qb + ((size_t)bb * SEQ + m0) * DIM);
    uint4* dst = (uint4*)sQ;
    for (int i = tid; i < BQ * DIM / 8; i += NTHREADS) dst[i] = src[i];
#endif
  } else {
    const float4* src = (const float4*)(Qf + ((size_t)bb * SEQ + m0) * DIM);
    uint32_t* dst = (uint32_t*)sQ;
    for (int i = tid; i < BQ * DIM / 4; i += NTHREADS) {
      float4 x = src[i];
      dst[2 * i]     = pack_bf16(x.x, x.y);
      dst[2 * i + 1] = pack_bf16(x.z, x.w);
    }
  }
  if (tid < BQ) { sM[tid] = -3.0e38f; sL[tid] = 0.0f; }

  v8f oacc[2][4];
  #pragma unroll
  for (int mt = 0; mt < 2; ++mt)
    #pragma unroll
    for (int dt = 0; dt < 4; ++dt)
      #pragma unroll
      for (int r = 0; r < 8; ++r) oacc[mt][dt][r] = 0.0f;

  for (int n0 = 0; n0 < SEQ; n0 += BK) {
    __syncthreads();                         // protect sKV/sS from previous iter

    // ---- stream KV tile into LDS ----
    if (BF16IN) {
#if HAVE_TDM
      if (wave == 0) {
        tdm_load_tile(Kb + ((size_t)bb * SEQ + n0) * DIM, sKV, BK, DIM);
        __builtin_amdgcn_s_wait_tensorcnt(0);
      }
#else
      const uint4* src = (const uint4*)(Kb + ((size_t)bb * SEQ + n0) * DIM);
      uint4* dst = (uint4*)sKV;
      for (int i = tid; i < BK * DIM / 8; i += NTHREADS) dst[i] = src[i];
#endif
    } else {
      const float4* src = (const float4*)(Kf + ((size_t)bb * SEQ + n0) * DIM);
      uint32_t* dst = (uint32_t*)sKV;
      for (int i = tid; i < BK * DIM / 4; i += NTHREADS) {
        float4 x = src[i];
        dst[2 * i]     = pack_bf16(x.x, x.y);
        dst[2 * i + 1] = pack_bf16(x.z, x.w);
      }
    }
    sS[tid] = 0.0f; sS[tid + NTHREADS] = 0.0f;
    __syncthreads();

    // ---- partial scores: one quadrant per wave, 256-wide D-slice ----
    {
      v8f acc;
      #pragma unroll
      for (int r = 0; r < 8; ++r) acc[r] = 0.0f;
      #pragma unroll
      for (int kk = 0; kk < SDSLICE / 32; ++kk) {
        const int c0 = sDs0 + kk * 32;
        v16bf a = load_a_frag (&sQ [sQuadM * DIM + c0], DIM, lane);
        v16bf b = load_b_trans(&sKV[sQuadN * DIM + c0], DIM, lane);
        acc = wmma_bf16(a, b, acc);
      }
      #pragma unroll
      for (int r = 0; r < 8; ++r)
        atomicAdd(&sS[(sQuadM + hh + r) * BK + sQuadN + cIn], acc[r]);
    }
    __syncthreads();

    // ---- online softmax: all 512 threads, 16 threads/row, shfl reductions ----
    {
      const float scale = 0.03125f;          // 1/sqrt(1024)
      const int row = tid >> 4;
      const int c0  = (tid & 15) * 2;
      const float* srow = &sS[row * BK + c0];
      float v0 = srow[0] * scale, v1 = srow[1] * scale;
      float mx = fmaxf(v0, v1);
      #pragma unroll
      for (int off = 1; off < 16; off <<= 1) mx = fmaxf(mx, __shfl_xor(mx, off, 16));
      const float mold = sM[row];
      const float mnew = fmaxf(mold, mx);
      const float p0 = __expf(v0 - mnew), p1 = __expf(v1 - mnew);
      *(uint32_t*)&sP[row * BK + c0] = pack_bf16(p0, p1);
      float sum = p0 + p1;
      #pragma unroll
      for (int off = 1; off < 16; off <<= 1) sum += __shfl_xor(sum, off, 16);
      if ((tid & 15) == 0) {
        const float corr = __expf(mold - mnew);
        sL[row] = sL[row] * corr + sum;
        sM[row] = mnew;
        sC[row] = corr;
      }
    }
    __syncthreads();

    // ---- rescale accumulator, then O += P * V on this wave's D-slice ----
    #pragma unroll
    for (int mt = 0; mt < 2; ++mt) {
      float corr[8];
      #pragma unroll
      for (int r = 0; r < 8; ++r) corr[r] = sC[mt * 16 + hh + r];
      #pragma unroll
      for (int dt = 0; dt < 4; ++dt)
        #pragma unroll
        for (int r = 0; r < 8; ++r) oacc[mt][dt][r] *= corr[r];

      v16bf pa = load_a_frag(&sP[mt * 16 * BK], BK, lane);
      #pragma unroll
      for (int dt = 0; dt < 4; ++dt) {
        v16bf vb = load_b_direct(&sKV[ds0 + dt * 16], DIM, lane);
        oacc[mt][dt] = wmma_bf16(pa, vb, oacc[mt][dt]);
      }
    }
  }
  __syncthreads();

  // ---- epilogue: Out = 0.5*Q + 0.5 * O / l ----
  #pragma unroll
  for (int mt = 0; mt < 2; ++mt) {
    float invl[8];
    #pragma unroll
    for (int r = 0; r < 8; ++r) invl[r] = 0.5f / sL[mt * 16 + hh + r];
    #pragma unroll
    for (int dt = 0; dt < 4; ++dt) {
      const int d = ds0 + dt * 16 + cIn;
      #pragma unroll
      for (int r = 0; r < 8; ++r) {
        const int row = mt * 16 + hh + r;
        const size_t idx = ((size_t)bb * SEQ + m0 + row) * (size_t)DIM + d;
        Og[idx] = 0.5f * Qf[idx] + oacc[mt][dt][r] * invl[r];
      }
    }
  }
}

extern "C" void kernel_launch(void* const* d_in, const int* in_sizes, int n_in,
                              void* d_out, int out_size, void* d_ws, size_t ws_size,
                              hipStream_t stream) {
  (void)in_sizes; (void)n_in; (void)out_size;
  // setup_inputs order: [0]=layer_key (int scalar), [1]=input1, [2]=input2
  const float* in1 = (const float*)d_in[1];
  const float* in2 = (const float*)d_in[2];
  float* out = (float*)d_out;
  const size_t N = (size_t)BATCH * SEQ * DIM;

  dim3 grid(SEQ / BQ, BATCH);
  if (ws_size >= 2 * N * sizeof(unsigned short)) {
    unsigned short* w1 = (unsigned short*)d_ws;
    unsigned short* w2 = w1 + N;
    convert_bf16<<<2048, 256, 0, stream>>>(in1, w1, (int)(N / 4));
    convert_bf16<<<2048, 256, 0, stream>>>(in2, w2, (int)(N / 4));
    cross_attn<true><<<grid, NTHREADS, 0, stream>>>(w1, w2, in1, in2, out);
    cross_attn<true><<<grid, NTHREADS, 0, stream>>>(w2, w1, in2, in1, out + N);
  } else {
    cross_attn<false><<<grid, NTHREADS, 0, stream>>>(nullptr, nullptr, in1, in2, out);
    cross_attn<false><<<grid, NTHREADS, 0, stream>>>(nullptr, nullptr, in2, in1, out + N);
  }
}
